// SelfAttention_7026566496383
// MI455X (gfx1250) — matrix operand order repaired and checked by reading, not compiled, not run on hardware
//
#include <hip/hip_runtime.h>
#include <hip/hip_bf16.h>

// S=2048, B=2, E=1024, H=16, D=64
#define S_LEN 2048
#define B_SZ  2
#define E_SZ  1024
#define H_CNT 16
#define D_SZ  64
#define M_ROWS (S_LEN * B_SZ)   // 4096
#define N_QKV  (3 * E_SZ)       // 3072

typedef __attribute__((ext_vector_type(16))) __bf16         v16bf;
typedef __attribute__((ext_vector_type(8)))  float          v8f;
typedef __attribute__((ext_vector_type(8)))  unsigned short v8u16;
typedef __attribute__((ext_vector_type(4)))  unsigned int   u32x4;
typedef __attribute__((ext_vector_type(8)))  int            i32x8;
typedef __attribute__((ext_vector_type(4)))  int            i32x4;

union Frag {
    v16bf bf;
    struct { v8u16 lo, hi; } p;
};

__device__ __forceinline__ unsigned short f2bf(float f) {
    unsigned int u = __float_as_uint(f);
    u += 0x7FFFu + ((u >> 16) & 1u);   // round-to-nearest-even
    return (unsigned short)(u >> 16);
}

// ---- Tensor Data Mover: 2D tile (global -> LDS), data_size = 2 bytes ----
// D# built per CDNA5 ISA 8.3/8.4: group0 {count=1, lds_addr, global_addr, type=2},
// group1 {data_size=1, tensor_dim0/1, tile_dim0/1, tensor_dim0_stride}, groups 2/3 zero (2D).
// This toolchain exposes the 6-arg builtin: (g0, g1, g2, g3, g_extra, cpol).
__device__ __forceinline__ void tdm_load_2d(unsigned int lds_addr,
                                            unsigned long long gaddr,
                                            unsigned int tensor_w,   // dim0 length (elements)
                                            unsigned int tensor_h,   // dim1 length (rows)
                                            unsigned int tile_w,     // tile dim0 (elements)
                                            unsigned int tile_h,     // tile dim1 (rows)
                                            unsigned long long stride0) // row stride (elements)
{
    u32x4 g0;
    g0[0] = 1u;                                             // count=1, user D#
    g0[1] = lds_addr;                                       // LDS byte address
    g0[2] = (unsigned int)gaddr;                            // global addr [31:0]
    g0[3] = (unsigned int)((gaddr >> 32) & 0x01FFFFFFull)   // global addr [56:32]
          | (2u << 30);                                     // type=2 ("image")

    unsigned long long q0 = (1ull << 16)                                   // data_size=1 -> 2B
                          | ((unsigned long long)(tensor_w & 0xFFFFu) << 48);
    unsigned long long q1 = ((unsigned long long)(tensor_w >> 16) & 0xFFFFull)
                          | ((unsigned long long)tensor_h << 16)
                          | ((unsigned long long)(tile_w & 0xFFFFu) << 48);
    unsigned long long q2 = (unsigned long long)(tile_h & 0xFFFFu)
                          | ((stride0 & 0xFFFFFFFFull) << 32);
    unsigned long long q3 = (stride0 >> 32) & 0xFFFFull;    // tensor_dim1_stride = 0 (2D)

    i32x8 g1;
    g1[0] = (int)q0; g1[1] = (int)(q0 >> 32);
    g1[2] = (int)q1; g1[3] = (int)(q1 >> 32);
    g1[4] = (int)q2; g1[5] = (int)(q2 >> 32);
    g1[6] = (int)q3; g1[7] = (int)(q3 >> 32);

    i32x4 z4 = {0, 0, 0, 0};
    i32x8 z8 = {0, 0, 0, 0, 0, 0, 0, 0};
    __builtin_amdgcn_tensor_load_to_lds(g0, g1, z4, z4, z8, 0);
}

// ---------------- f32 -> bf16 convert ----------------
__global__ void cvt_bf16_kernel(const float* __restrict__ in,
                                unsigned short* __restrict__ out, int n) {
    int i = blockIdx.x * blockDim.x + threadIdx.x;
    if (i < n) out[i] = f2bf(in[i]);
}

// ---------------- fused QKV GEMM with TDM + LDS staging --------------------
// Block tile: 64(M) x 128(N), K stepped by 64. 8 waves, each owns 32x32.
// A [4096,1024] bf16, W [3072,1024] bf16 (both K-contiguous), out [4096,3072] bf16.
__global__ __launch_bounds__(256) void qkv_gemm_kernel(
    const unsigned short* __restrict__ A,
    const unsigned short* __restrict__ W,
    const float* __restrict__ bias,
    unsigned short* __restrict__ out)
{
    __shared__ unsigned short ldsA[64 * 64];    // 8 KB
    __shared__ unsigned short ldsW[128 * 64];   // 16 KB

    const int lane = threadIdx.x & 31;
    const int wave = threadIdx.x >> 5;
    const int half = lane >> 4;
    const int ln   = lane & 15;
    const int wm   = wave >> 2;          // 0..1 -> M offset 32*wm
    const int wn   = wave & 3;           // 0..3 -> N offset 32*wn
    const int bm   = blockIdx.y * 64;    // block M origin
    const int bn   = blockIdx.x * 128;   // block N origin

    const unsigned int ldsA_addr = (unsigned int)(size_t)(void*)ldsA;
    const unsigned int ldsW_addr = (unsigned int)(size_t)(void*)ldsW;
    const unsigned long long aBase = (unsigned long long)(size_t)(A + (size_t)bm * E_SZ);
    const unsigned long long wBase = (unsigned long long)(size_t)(W + (size_t)bn * E_SZ);

    v8f acc[2][2] = {{{}, {}}, {{}, {}}};

    for (int k0 = 0; k0 < E_SZ; k0 += 64) {
        if (wave == 0) {
            tdm_load_2d(ldsA_addr, aBase + (unsigned long long)k0 * 2,
                        E_SZ, M_ROWS, 64, 64, E_SZ);
            tdm_load_2d(ldsW_addr, wBase + (unsigned long long)k0 * 2,
                        E_SZ, N_QKV, 64, 128, E_SZ);
            __builtin_amdgcn_s_wait_tensorcnt(0);
        }
        __syncthreads();

#pragma unroll
        for (int kc = 0; kc < 2; ++kc) {          // two 32-deep K chunks
            Frag a[2], b[2];
#pragma unroll
            for (int mi = 0; mi < 2; ++mi) {
                const v8u16* ap = (const v8u16*)&ldsA[(wm * 32 + mi * 16 + ln) * 64];
                a[mi].p.lo = ap[kc * 4 + half];       // K = 32*kc + 8*half  .. +7
                a[mi].p.hi = ap[kc * 4 + 2 + half];   // K = 32*kc +16+8*half.. +7
            }
#pragma unroll
            for (int ni = 0; ni < 2; ++ni) {
                const v8u16* wp = (const v8u16*)&ldsW[(wn * 32 + ni * 16 + ln) * 64];
                b[ni].p.lo = wp[kc * 4 + half];       // K = 32*kc + 8*half  .. +7
                b[ni].p.hi = wp[kc * 4 + 2 + half];   // K = 32*kc +16+8*half.. +7
            }
#pragma unroll
            for (int mi = 0; mi < 2; ++mi)
#pragma unroll
                for (int ni = 0; ni < 2; ++ni)
                    acc[mi][ni] = __builtin_amdgcn_wmma_f32_16x16x32_bf16(
                        false, a[mi].bf, false, b[ni].bf, (short)0, acc[mi][ni], false, false);
        }
        __syncthreads();   // protect LDS before next TDM overwrite
    }

#pragma unroll
    for (int ni = 0; ni < 2; ++ni) {
        const int f = bn + wn * 32 + ni * 16 + ln;
        const float bv = bias[f];
        const float sc = (f < E_SZ) ? 0.125f : 1.0f;    // fold 1/sqrt(D) into Q
#pragma unroll
        for (int mi = 0; mi < 2; ++mi) {
#pragma unroll
            for (int r = 0; r < 8; ++r) {
                const int m = bm + wm * 32 + mi * 16 + r + 8 * half;
                out[(size_t)m * N_QKV + f] = f2bf((acc[mi][ni][r] + bv) * sc);
            }
        }
    }
}

// ---------------- V transpose: vt[b,h,d,t] = qkv[t,b, 2E + h*64 + d] ------
__global__ void transpose_v_kernel(const unsigned short* __restrict__ qkv,
                                   unsigned short* __restrict__ vt, int total) {
    int i = blockIdx.x * blockDim.x + threadIdx.x;
    if (i >= total) return;
    int t = i & (S_LEN - 1);
    int r = i >> 11;
    int d = r & (D_SZ - 1); r >>= 6;
    int h = r & (H_CNT - 1);
    int b = r >> 4;
    vt[i] = qkv[(size_t)(t * B_SZ + b) * N_QKV + 2 * E_SZ + h * D_SZ + d];
}

// ---------------- flash attention: one wave per 16 query rows --------------
__global__ __launch_bounds__(256) void attn_kernel(
    const unsigned short* __restrict__ qkv,   // [4096,3072] bf16 (q scaled)
    const unsigned short* __restrict__ vt,    // [B,H,D,S] bf16
    float* __restrict__ out)                  // [S,B,E] f32
{
    __shared__ unsigned short ldsP[8][16 * 32];   // per-wave P staging (C->A layout)

    const int lane = threadIdx.x & 31;
    const int wave = threadIdx.x >> 5;
    const int half = lane >> 4;
    const int ln   = lane & 15;
    const int qt = blockIdx.x * 8 + wave;   // query tile (16 rows)
    const int h  = blockIdx.y;
    const int b  = blockIdx.z;

    // Q A-fragments (16x64 = two 16x32 chunks), kept in registers all loop
    Frag qf[2];
    {
        const int s = qt * 16 + ln;
        const v8u16* qp = (const v8u16*)(qkv + (size_t)(s * B_SZ + b) * N_QKV + h * D_SZ);
#pragma unroll
        for (int kc = 0; kc < 2; ++kc) {
            qf[kc].p.lo = qp[kc * 4 + half];
            qf[kc].p.hi = qp[kc * 4 + 2 + half];
        }
    }

    const unsigned short* kbase = qkv + (size_t)b * N_QKV + E_SZ + h * D_SZ;
    const unsigned short* vbase = vt + (size_t)((b * H_CNT + h) * D_SZ) * S_LEN;

    v8f ctx[4] = {{}, {}, {}, {}};
    float mrow[8], srow[8];
#pragma unroll
    for (int r = 0; r < 8; ++r) { mrow[r] = -1e30f; srow[r] = 0.0f; }

    for (int t0 = 0; t0 < S_LEN; t0 += 32) {
        // ---- scores: 16 queries x 32 keys (two 16x16 C tiles, 4 WMMAs) ----
        v8f sc[2];
#pragma unroll
        for (int c = 0; c < 2; ++c) {
            const int krow = t0 + c * 16 + ln;                    // key row = column n
            const v8u16* kp = (const v8u16*)(kbase + (size_t)krow * (B_SZ * N_QKV));
            v8f accv = {};
#pragma unroll
            for (int kc = 0; kc < 2; ++kc) {
                Frag kb;
                kb.p.lo = kp[kc * 4 + half];
                kb.p.hi = kp[kc * 4 + 2 + half];
                accv = __builtin_amdgcn_wmma_f32_16x16x32_bf16(
                    false, qf[kc].bf, false, kb.bf, (short)0, accv, false, false);
            }
            sc[c] = accv;
        }

        // ---- online softmax over the 32 new keys ----
#pragma unroll
        for (int r = 0; r < 8; ++r) {
            float mloc = fmaxf(sc[0][r], sc[1][r]);
#pragma unroll
            for (int off = 1; off < 16; off <<= 1)
                mloc = fmaxf(mloc, __shfl_xor(mloc, off));        // 16-lane half reduce
            const float mn   = fmaxf(mrow[r], mloc);
            const float corr = __expf(mrow[r] - mn);
            mrow[r] = mn;
            const float p0 = __expf(sc[0][r] - mn);
            const float p1 = __expf(sc[1][r] - mn);
            float ls = p0 + p1;
#pragma unroll
            for (int off = 1; off < 16; off <<= 1)
                ls += __shfl_xor(ls, off);
            srow[r] = srow[r] * corr + ls;
            ctx[0][r] *= corr; ctx[1][r] *= corr;
            ctx[2][r] *= corr; ctx[3][r] *= corr;
            // stage P (C layout) into LDS row-major 16x32
            const int m = r + 8 * half;
            ldsP[wave][m * 32 + ln]      = f2bf(p0);
            ldsP[wave][m * 32 + 16 + ln] = f2bf(p1);
        }
        asm volatile("s_wait_dscnt 0" ::: "memory");

        // reload P as an A-fragment (16x32)
        Frag pf;
        {
            const v8u16* rowp = (const v8u16*)&ldsP[wave][ln * 32];
            pf.p.lo = rowp[half];       // K = 8*half .. +7
            pf.p.hi = rowp[2 + half];   // K = 16+8*half .. +7
        }

        // ---- ctx += P(16x32) x V(32x64): 4 WMMAs over d-tiles ----
#pragma unroll
        for (int dt = 0; dt < 4; ++dt) {
            Frag vf;
            const v8u16* vp = (const v8u16*)(vbase + (size_t)(dt * 16 + ln) * S_LEN + t0);
            vf.p.lo = vp[half];
            vf.p.hi = vp[2 + half];
            ctx[dt] = __builtin_amdgcn_wmma_f32_16x16x32_bf16(
                false, pf.bf, false, vf.bf, (short)0, ctx[dt], false, false);
        }
    }

    // ---- normalize and store f32 output ----
    float inv[8];
#pragma unroll
    for (int r = 0; r < 8; ++r) inv[r] = 1.0f / srow[r];
#pragma unroll
    for (int dt = 0; dt < 4; ++dt) {
#pragma unroll
        for (int r = 0; r < 8; ++r) {
            const int s = qt * 16 + r + 8 * half;
            const int e = h * D_SZ + dt * 16 + ln;
            out[(size_t)(s * B_SZ + b) * E_SZ + e] = ctx[dt][r] * inv[r];
        }
    }
}

// ---------------- host launch ----------------
extern "C" void kernel_launch(void* const* d_in, const int* in_sizes, int n_in,
                              void* d_out, int out_size, void* d_ws, size_t ws_size,
                              hipStream_t stream) {
    const float* x    = (const float*)d_in[0];   // [S,B,E]
    const float* Wqkv = (const float*)d_in[1];   // [3E,E]
    const float* bqkv = (const float*)d_in[2];   // [3E]
    float* out = (float*)d_out;                  // [S,B,E]

    unsigned short* xb   = (unsigned short*)d_ws;                 // 4096*1024
    unsigned short* wb   = xb   + (size_t)M_ROWS * E_SZ;          // 3072*1024
    unsigned short* qkvb = wb   + (size_t)N_QKV  * E_SZ;          // 4096*3072
    unsigned short* vtb  = qkvb + (size_t)M_ROWS * N_QKV;         // 2*16*64*2048

    const int nx = M_ROWS * E_SZ;
    const int nw = N_QKV * E_SZ;
    const int nv = B_SZ * H_CNT * D_SZ * S_LEN;

    cvt_bf16_kernel<<<(nx + 255) / 256, 256, 0, stream>>>(x, xb, nx);
    cvt_bf16_kernel<<<(nw + 255) / 256, 256, 0, stream>>>(Wqkv, wb, nw);

    // grid: N/128=24 x M/64=64, 8 waves per block, TDM-staged LDS tiles
    qkv_gemm_kernel<<<dim3(N_QKV / 128, M_ROWS / 64), 256, 0, stream>>>(xb, wb, bqkv, qkvb);

    transpose_v_kernel<<<(nv + 255) / 256, 256, 0, stream>>>(qkvb, vtb, nv);

    // grid: (S/16/8)=16 query-tile groups x H x B, 8 waves per block
    attn_kernel<<<dim3(S_LEN / 128, H_CNT, B_SZ), 256, 0, stream>>>(qkvb, vtb, out);
}
